// AttentionBlock_17214228922517
// MI455X (gfx1250) — matrix operand order repaired and checked by reading, not compiled, not run on hardware
//
#include <hip/hip_runtime.h>
#include <hip/hip_bf16.h>

// ---------------------------------------------------------------------------
// AttentionBlock for MI455X (gfx1250, wave32, WMMA).
// All GEMMs use v_wmma_f32_16x16x32_f16 (f16 in, f32 accumulate).
// Working set (~145MB) fits in the 192MB L2, so GEMMs are compute-bound.
// ---------------------------------------------------------------------------

typedef _Float16 h4  __attribute__((ext_vector_type(4)));
typedef _Float16 h8  __attribute__((ext_vector_type(8)));
typedef _Float16 v16h __attribute__((ext_vector_type(16)));
typedef float    v8f __attribute__((ext_vector_type(8)));

#define B_SZ   32
#define C_SZ   256
#define HW_SZ  1024
#define OC3    768
#define LDSK   48   // padded LDS row stride in f16 (96B, 16B-aligned rows)

__device__ __forceinline__ v16h cat8(h8 lo, h8 hi) {
  return __builtin_shufflevector(lo, hi, 0,1,2,3,4,5,6,7,8,9,10,11,12,13,14,15);
}

// ---------------------------------------------------------------------------
// Shared GEMM core: C(128x128) += A(128xK) * B(Kx128)
// A: row-major (m,k) with leading dim lda, pre-offset to tile origin (m_blk,0)
// B: row-major (k,n) with leading dim ldb, pre-offset to tile origin (0,n_blk)
// 256 threads = 8 waves; wave (wm,wn) owns 4 (M) x 2 (N) 16x16 accumulators.
// ---------------------------------------------------------------------------
__device__ __forceinline__ void gemm_core(
    const _Float16* __restrict__ A, int lda,
    const _Float16* __restrict__ B, int ldb,
    int K, _Float16* lds, v8f acc[4][2])
{
  _Float16* lA = lds;               // [128][LDSK]  (m-major, k inner)
  _Float16* lB = lds + 128 * LDSK;  // [128][LDSK]  (n-major, k inner)

  const int tid  = threadIdx.x;
  const int lane = tid & 31;
  const int wv   = tid >> 5;
  const int wm   = (wv & 1) * 64;   // wave M offset
  const int wn   = (wv >> 1) * 32;  // wave N offset
  const int lm   = lane & 15;
  const int kboA = (lane < 16) ? 0 : 8;   // A fragment K base (16-bit layout)
  const int kboB = (lane < 16) ? 0 : 8;   // B fragment K base

  // staging roles
  const int arow  = tid >> 1;          // 0..127
  const int acol  = (tid & 1) * 16;    // 0 or 16
  const int kpair = (tid >> 4) * 2;    // 0,2,..,30 (two consecutive k rows)
  const int ncol  = (tid & 15) * 8;    // 0..120 (8-wide n segment)

  const v8f zero = {0.f,0.f,0.f,0.f,0.f,0.f,0.f,0.f};
  #pragma unroll
  for (int i = 0; i < 4; ++i)
    #pragma unroll
    for (int j = 0; j < 2; ++j) acc[i][j] = zero;

  for (int k0 = 0; k0 < K; k0 += 32) {
    // ---- stage A tile 128x32 (straight copy, 2x b128 per thread)
    const h8* ga = (const h8*)(A + (size_t)arow * lda + k0 + acol);
    h8 a0 = ga[0];
    h8 a1 = ga[1];
    *(h8*)(lA + arow * LDSK + acol)     = a0;
    *(h8*)(lA + arow * LDSK + acol + 8) = a1;

    // ---- stage B tile 32x128, transposed into LDS as [n][k].
    // Two consecutive k-rows per thread -> k-adjacent pairs are contiguous
    // in LDS, so writes are packed 2xf16 ds_store_b32 (8 per thread).
    h8 b0 = *(const h8*)(B + (size_t)(k0 + kpair)     * ldb + ncol);
    h8 b1 = *(const h8*)(B + (size_t)(k0 + kpair + 1) * ldb + ncol);
    #pragma unroll
    for (int e = 0; e < 8; ++e) {
      union { _Float16 h[2]; unsigned int u; } pk;
      pk.h[0] = b0[e];
      pk.h[1] = b1[e];
      *(unsigned int*)(lB + (ncol + e) * LDSK + kpair) = pk.u;
    }

    // prefetch next K tile into near caches (global_prefetch_b8, WGP scope)
    if (k0 + 32 < K) {
      __builtin_prefetch(A + (size_t)arow * lda + k0 + 32 + acol, 0, 3);
      __builtin_prefetch(B + (size_t)(k0 + 32 + kpair) * ldb + ncol, 0, 3);
    }

    __syncthreads();

    // ---- B fragments for this wave's 2 N-tiles (2x ds_load_b128 each)
    v16h bfrag[2];
    #pragma unroll
    for (int j = 0; j < 2; ++j) {
      const _Float16* p = lB + (wn + j * 16 + lm) * LDSK + kboB;
      bfrag[j] = cat8(*(const h8*)p, *(const h8*)(p + 16));
    }

    // ---- A fragments + 8 WMMAs
    #pragma unroll
    for (int i = 0; i < 4; ++i) {
      const _Float16* p = lA + (wm + i * 16 + lm) * LDSK + kboA;
      v16h afrag = cat8(*(const h8*)p, *(const h8*)(p + 16));
      #pragma unroll
      for (int j = 0; j < 2; ++j) {
        acc[i][j] = __builtin_amdgcn_wmma_f32_16x16x32_f16(
            false, afrag, false, bfrag[j], (short)0, acc[i][j], false, false);
      }
    }
    __syncthreads();
  }
}

// ---------------------------------------------------------------------------
// Kernel 0: weight prep — f32 -> f16 with transpose to (K=c, N=o)
// ---------------------------------------------------------------------------
__global__ __launch_bounds__(256) void prep_w_kernel(
    const float* __restrict__ w_qkv, const float* __restrict__ w_out,
    _Float16* __restrict__ wqT, _Float16* __restrict__ woT)
{
  int idx = blockIdx.x * 256 + threadIdx.x;
  if (idx < OC3 * C_SZ) {
    int o = idx >> 8, c = idx & 255;
    wqT[c * OC3 + o] = (_Float16)w_qkv[idx];
  } else {
    int k = idx - OC3 * C_SZ;
    int o = k >> 8, c = k & 255;
    woT[c * C_SZ + o] = (_Float16)w_out[k];
  }
}

// ---------------------------------------------------------------------------
// Kernel 1: GroupNorm. One block per (batch, group). Writes xn as (b, s, c)
// f16 via LDS 32x32 transpose tiles so reads and writes are both coalesced.
// ---------------------------------------------------------------------------
__global__ __launch_bounds__(256) void groupnorm_kernel(
    const float* __restrict__ x, const float* __restrict__ gw,
    const float* __restrict__ gb, _Float16* __restrict__ xn)
{
  const int bg = blockIdx.x;
  const int b = bg >> 3, g = bg & 7;
  const float* xp = x + (size_t)(b * C_SZ + g * 32) * HW_SZ;
  const int tid = threadIdx.x;

  float s = 0.f, ss = 0.f;
  for (int e = tid; e < 32 * HW_SZ; e += 256) {
    float v = xp[e];
    s += v; ss += v * v;
  }
  __shared__ float red[512];
  __shared__ float mean_s, inv_s;
  red[tid] = s; red[256 + tid] = ss;
  __syncthreads();
  for (int off = 128; off > 0; off >>= 1) {
    if (tid < off) {
      red[tid]       += red[tid + off];
      red[256 + tid] += red[256 + tid + off];
    }
    __syncthreads();
  }
  if (tid == 0) {
    float mean = red[0] * (1.f / 32768.f);
    float var  = red[256] * (1.f / 32768.f) - mean * mean;
    mean_s = mean;
    inv_s  = rsqrtf(var + 1e-5f);
  }
  __syncthreads();
  const float mean = mean_s, inv = inv_s;

  __shared__ _Float16 tile[32][33];
  const int rc = tid >> 3;          // channel (read) / s-row (write)
  const int q4 = (tid & 7) * 4;     // 4-wide column
  const float w_c = gw[g * 32 + rc] * inv;
  const float b_c = gb[g * 32 + rc] - mean * w_c;

  for (int s0 = 0; s0 < HW_SZ; s0 += 32) {
    const float4 vx = *(const float4*)(xp + (size_t)rc * HW_SZ + s0 + q4);
    tile[q4 + 0][rc] = (_Float16)(vx.x * w_c + b_c);
    tile[q4 + 1][rc] = (_Float16)(vx.y * w_c + b_c);
    tile[q4 + 2][rc] = (_Float16)(vx.z * w_c + b_c);
    tile[q4 + 3][rc] = (_Float16)(vx.w * w_c + b_c);
    __syncthreads();
    _Float16* dst = xn + ((size_t)(b * HW_SZ + s0 + rc)) * C_SZ + g * 32 + q4;
    h4 o;
    o[0] = tile[rc][q4 + 0];
    o[1] = tile[rc][q4 + 1];
    o[2] = tile[rc][q4 + 2];
    o[3] = tile[rc][q4 + 3];
    *(h4*)dst = o;
    __syncthreads();
  }
}

// ---------------------------------------------------------------------------
// Kernel 2: QKV GEMM. qkv[b,s,o] = xn[b,s,:] . wqT[:,o] + bias.
// q,v stored (b,s,c); k stored (b,c,s) so QK^T and attn*V need no transposes.
// ---------------------------------------------------------------------------
__global__ __launch_bounds__(256) void gemm_qkv_kernel(
    const _Float16* __restrict__ xn, const _Float16* __restrict__ wqT,
    const float* __restrict__ b_qkv,
    _Float16* __restrict__ qT, _Float16* __restrict__ kbuf,
    _Float16* __restrict__ vT)
{
  __shared__ _Float16 lds[2 * 128 * LDSK];
  const int n_blk = blockIdx.x * 128;
  const int m_blk = blockIdx.y * 128;
  const int b     = blockIdx.z;
  const _Float16* A = xn + (size_t)b * HW_SZ * C_SZ + (size_t)m_blk * C_SZ;
  const _Float16* B = wqT + n_blk;
  v8f acc[4][2];
  gemm_core(A, C_SZ, B, OC3, C_SZ, lds, acc);

  const int lane = threadIdx.x & 31;
  const int wv = threadIdx.x >> 5;
  const int wm = (wv & 1) * 64, wn = (wv >> 1) * 32;
  const int mb = (lane >> 4) * 8, lm = lane & 15;
  #pragma unroll
  for (int j = 0; j < 2; ++j) {
    const int N = n_blk + wn + j * 16 + lm;
    const float bias = b_qkv[N];
    #pragma unroll
    for (int i = 0; i < 4; ++i) {
      #pragma unroll
      for (int r = 0; r < 8; ++r) {
        const int M = m_blk + wm + i * 16 + mb + r;
        const float v = acc[i][j][r] + bias;
        if (N < 256)      qT[((size_t)b * HW_SZ + M) * C_SZ + N] = (_Float16)v;
        else if (N < 512) kbuf[((size_t)b * C_SZ + (N - 256)) * HW_SZ + M] = (_Float16)v;
        else              vT[((size_t)b * HW_SZ + M) * C_SZ + (N - 512)] = (_Float16)v;
      }
    }
  }
}

// ---------------------------------------------------------------------------
// Kernel 3: logits = (q^T k) * 1/sqrt(C), stored f16 (b, s, t)
// ---------------------------------------------------------------------------
__global__ __launch_bounds__(256) void gemm_qk_kernel(
    const _Float16* __restrict__ qT, const _Float16* __restrict__ kbuf,
    _Float16* __restrict__ att)
{
  __shared__ _Float16 lds[2 * 128 * LDSK];
  const int n_blk = blockIdx.x * 128;
  const int m_blk = blockIdx.y * 128;
  const int b     = blockIdx.z;
  const _Float16* A = qT + (size_t)b * HW_SZ * C_SZ + (size_t)m_blk * C_SZ;
  const _Float16* B = kbuf + (size_t)b * C_SZ * HW_SZ + n_blk;
  v8f acc[4][2];
  gemm_core(A, C_SZ, B, HW_SZ, C_SZ, lds, acc);

  const int lane = threadIdx.x & 31;
  const int wv = threadIdx.x >> 5;
  const int wm = (wv & 1) * 64, wn = (wv >> 1) * 32;
  const int mb = (lane >> 4) * 8, lm = lane & 15;
  #pragma unroll
  for (int i = 0; i < 4; ++i)
    #pragma unroll
    for (int j = 0; j < 2; ++j)
      #pragma unroll
      for (int r = 0; r < 8; ++r) {
        const int M = m_blk + wm + i * 16 + mb + r;
        const int N = n_blk + wn + j * 16 + lm;
        att[((size_t)b * HW_SZ + M) * HW_SZ + N] = (_Float16)(acc[i][j][r] * 0.0625f);
      }
}

// ---------------------------------------------------------------------------
// Kernel 4: in-place row softmax on f16 attention (one block per (b, s) row)
// ---------------------------------------------------------------------------
__global__ __launch_bounds__(256) void softmax_kernel(_Float16* __restrict__ att)
{
  _Float16* p = att + ((size_t)(blockIdx.y * HW_SZ + blockIdx.x)) * HW_SZ;
  const int tid = threadIdx.x;

  h4 hv = *(const h4*)(p + tid * 4);
  float v[4];
  float m = -1e30f;
  #pragma unroll
  for (int i = 0; i < 4; ++i) { v[i] = (float)hv[i]; m = fmaxf(m, v[i]); }

  __shared__ float red[256];
  red[tid] = m;
  __syncthreads();
  for (int off = 128; off > 0; off >>= 1) {
    if (tid < off) red[tid] = fmaxf(red[tid], red[tid + off]);
    __syncthreads();
  }
  const float rmax = red[0];
  __syncthreads();

  float s = 0.f;
  #pragma unroll
  for (int i = 0; i < 4; ++i) { v[i] = __expf(v[i] - rmax); s += v[i]; }
  red[tid] = s;
  __syncthreads();
  for (int off = 128; off > 0; off >>= 1) {
    if (tid < off) red[tid] += red[tid + off];
    __syncthreads();
  }
  const float inv = 1.f / red[0];

  h4 o;
  #pragma unroll
  for (int i = 0; i < 4; ++i) o[i] = (_Float16)(v[i] * inv);
  *(h4*)(p + tid * 4) = o;
}

// ---------------------------------------------------------------------------
// Kernel 5: out_t[b,s,c] = attn[b,s,:] . vT[:,c]
// ---------------------------------------------------------------------------
__global__ __launch_bounds__(256) void gemm_av_kernel(
    const _Float16* __restrict__ att, const _Float16* __restrict__ vT,
    _Float16* __restrict__ outT)
{
  __shared__ _Float16 lds[2 * 128 * LDSK];
  const int n_blk = blockIdx.x * 128;
  const int m_blk = blockIdx.y * 128;
  const int b     = blockIdx.z;
  const _Float16* A = att + (size_t)b * HW_SZ * HW_SZ + (size_t)m_blk * HW_SZ;
  const _Float16* B = vT + (size_t)b * HW_SZ * C_SZ + n_blk;
  v8f acc[4][2];
  gemm_core(A, HW_SZ, B, C_SZ, HW_SZ, lds, acc);

  const int lane = threadIdx.x & 31;
  const int wv = threadIdx.x >> 5;
  const int wm = (wv & 1) * 64, wn = (wv >> 1) * 32;
  const int mb = (lane >> 4) * 8, lm = lane & 15;
  #pragma unroll
  for (int i = 0; i < 4; ++i)
    #pragma unroll
    for (int j = 0; j < 2; ++j)
      #pragma unroll
      for (int r = 0; r < 8; ++r) {
        const int M = m_blk + wm + i * 16 + mb + r;
        const int N = n_blk + wn + j * 16 + lm;
        outT[((size_t)b * HW_SZ + M) * C_SZ + N] = (_Float16)acc[i][j][r];
      }
}

// ---------------------------------------------------------------------------
// Kernel 6: final projection + bias + residual, f32 output (b, c, h, w)
// ---------------------------------------------------------------------------
__global__ __launch_bounds__(256) void gemm_proj_kernel(
    const _Float16* __restrict__ outT, const _Float16* __restrict__ woT,
    const float* __restrict__ b_out, const float* __restrict__ x,
    float* __restrict__ out)
{
  __shared__ _Float16 lds[2 * 128 * LDSK];
  const int n_blk = blockIdx.x * 128;
  const int m_blk = blockIdx.y * 128;
  const int b     = blockIdx.z;
  const _Float16* A = outT + (size_t)b * HW_SZ * C_SZ + (size_t)m_blk * C_SZ;
  const _Float16* B = woT + n_blk;
  v8f acc[4][2];
  gemm_core(A, C_SZ, B, C_SZ, C_SZ, lds, acc);

  const int lane = threadIdx.x & 31;
  const int wv = threadIdx.x >> 5;
  const int wm = (wv & 1) * 64, wn = (wv >> 1) * 32;
  const int mb = (lane >> 4) * 8, lm = lane & 15;
  #pragma unroll
  for (int j = 0; j < 2; ++j) {
    const int N = n_blk + wn + j * 16 + lm;
    const float bias = b_out[N];
    #pragma unroll
    for (int i = 0; i < 4; ++i) {
      #pragma unroll
      for (int r = 0; r < 8; ++r) {
        const int M = m_blk + wm + i * 16 + mb + r;
        const size_t idx = ((size_t)b * C_SZ + N) * HW_SZ + M;
        out[idx] = acc[i][j][r] + bias + x[idx];
      }
    }
  }
}

// ---------------------------------------------------------------------------
// Host launcher
// ---------------------------------------------------------------------------
extern "C" void kernel_launch(void* const* d_in, const int* in_sizes, int n_in,
                              void* d_out, int out_size, void* d_ws, size_t ws_size,
                              hipStream_t stream) {
  const float* x     = (const float*)d_in[0];
  const float* gnw   = (const float*)d_in[1];
  const float* gnb   = (const float*)d_in[2];
  const float* w_qkv = (const float*)d_in[3];
  const float* b_qkv = (const float*)d_in[4];
  const float* w_out = (const float*)d_in[5];
  const float* b_out = (const float*)d_in[6];
  float* out = (float*)d_out;

  char* ws = (char*)d_ws;
  size_t off = 0;
  auto alloc = [&](size_t bytes) -> void* {
    void* p = ws + off;
    off += (bytes + 255) & ~(size_t)255;
    return p;
  };
  const size_t act = (size_t)B_SZ * HW_SZ * C_SZ * sizeof(_Float16);  // 16MB
  _Float16* wqT  = (_Float16*)alloc((size_t)C_SZ * OC3 * sizeof(_Float16));
  _Float16* woT  = (_Float16*)alloc((size_t)C_SZ * C_SZ * sizeof(_Float16));
  _Float16* xn   = (_Float16*)alloc(act);
  _Float16* qT   = (_Float16*)alloc(act);
  _Float16* kbuf = (_Float16*)alloc(act);
  _Float16* vT   = (_Float16*)alloc(act);
  _Float16* outT = (_Float16*)alloc(act);
  _Float16* att  = (_Float16*)alloc((size_t)B_SZ * HW_SZ * HW_SZ * sizeof(_Float16)); // 64MB

  prep_w_kernel<<<(OC3 * C_SZ + C_SZ * C_SZ) / 256, 256, 0, stream>>>(w_qkv, w_out, wqT, woT);
  groupnorm_kernel<<<B_SZ * 8, 256, 0, stream>>>(x, gnw, gnb, xn);
  gemm_qkv_kernel<<<dim3(OC3 / 128, HW_SZ / 128, B_SZ), 256, 0, stream>>>(
      xn, wqT, b_qkv, qT, kbuf, vT);
  gemm_qk_kernel<<<dim3(HW_SZ / 128, HW_SZ / 128, B_SZ), 256, 0, stream>>>(qT, kbuf, att);
  softmax_kernel<<<dim3(HW_SZ, B_SZ), 256, 0, stream>>>(att);
  gemm_av_kernel<<<dim3(C_SZ / 128, HW_SZ / 128, B_SZ), 256, 0, stream>>>(att, vT, outT);
  gemm_proj_kernel<<<dim3(C_SZ / 128, HW_SZ / 128, B_SZ), 256, 0, stream>>>(
      outT, woT, b_out, x, out);
}
